// GAT_12584254177621
// MI455X (gfx1250) — matrix operand order repaired and checked by reading, not compiled, not run on hardware
//
#include <hip/hip_runtime.h>

// ---------------------------------------------------------------------------
// GAT (2-layer, PyG-style) for MI455X / gfx1250.
// Dense projections -> bf16 WMMA (v_wmma_f32_16x16x32_bf16, f32 accum),
//   W pre-packed into a contiguous-k B-fragment layout (global, L2-resident),
//   A tile staged to LDS via GLOBAL_LOAD_ASYNC_TO_LDS (double buffered).
// Graph ops -> L2-resident gathers + f32 global atomics.
// ---------------------------------------------------------------------------

typedef __attribute__((ext_vector_type(16))) __bf16 v16bf;
typedef __attribute__((ext_vector_type(8)))  float  v8f;
typedef __attribute__((ext_vector_type(4)))  int    i4v;

#if __has_builtin(__builtin_amdgcn_global_load_async_to_lds_b128)
#define GAT_ASYNC_LDS 1
typedef __attribute__((address_space(1))) i4v  gi4v;   // global int4
typedef __attribute__((address_space(3))) i4v  li4v;   // LDS int4
typedef __attribute__((address_space(1))) void gvoid;
typedef __attribute__((address_space(3))) void lvoid;
#endif

#define NN    30000
#define EE    480000
#define ETOT  (EE + NN)         // with self loops
#define FEAT  512
#define HCOL  512               // H*C for both layers (8*64)
#define HEADS 8
#define CDIM  64                // HID == NUM_CLASSES == 64
#define NEGS  0.2f

// ------------------------------- helpers -----------------------------------

__device__ __forceinline__ float drop_mask(unsigned int idx, unsigned int seed) {
    unsigned int x = idx * 2654435761u ^ seed;
    x ^= x >> 16; x *= 0x7feb352du;
    x ^= x >> 15; x *= 0x846ca68bu;
    x ^= x >> 16;
    float u = (float)(x >> 8) * (1.0f / 16777216.0f);
    return (u < 0.8f) ? 1.25f : 0.0f;   // keep p=0.8, scale 1/(1-0.2)
}

__device__ __forceinline__ float elu_f(float v) {
    return v > 0.0f ? v : (__expf(v) - 1.0f);
}

__device__ __forceinline__ void atomic_max_f32(float* addr, float val) {
    unsigned int* ua = (unsigned int*)addr;
    unsigned int cur = __float_as_uint(*addr);
    while (__uint_as_float(cur) < val) {
        unsigned int assumed = cur;
        cur = atomicCAS(ua, assumed, __float_as_uint(val));
        if (cur == assumed) break;
    }
}

// --------------------------- elementwise kernels ---------------------------

__global__ void gat_fill_f32(float* __restrict__ p, float v, int n) {
    int i = blockIdx.x * blockDim.x + threadIdx.x;
    if (i < n) p[i] = v;
}

// h = bf16( elu( dropout(x) ) )
__global__ void gat_prep_x(const float* __restrict__ x, __bf16* __restrict__ hb,
                           int n, unsigned int seed) {
    int i = blockIdx.x * blockDim.x + threadIdx.x;
    if (i >= n) return;
    float v = x[i] * drop_mask((unsigned)i, seed);
    hb[i] = (__bf16)elu_f(v);
}

// Pack W (f32 [512][512] row-major, K x N) into a contiguous-k 16-bit B-matrix
// fragment layout, bf16. Chunk order: [kslab][nblk][lane][elem] where each
// lane owns 16 contiguous bf16 (= two b128 loads in the GEMM).
//   lane L (0..31): col = nblk*16 + (L&15), kbase = (L>>4)*16
//   elem v (0..15): k = kslab*32 + kbase + v
__global__ void gat_pack_w(const float* __restrict__ w, __bf16* __restrict__ wf, int n) {
    int i = blockIdx.x * blockDim.x + threadIdx.x;
    if (i >= n) return;                      // n = 512*512
    int v     = i & 15;
    int L     = (i >> 4) & 31;
    int nblk  = (i >> 9) & 31;
    int kslab = i >> 14;
    int col = nblk * 16 + (L & 15);
    int k   = kslab * 32 + (L >> 4) * 16 + v;
    wf[i] = (__bf16)w[k * HCOL + col];
}

// between layers: h2 = bf16( elu( dropout(acc1 + b1) ) )
__global__ void gat_mid(const float* __restrict__ acc, const float* __restrict__ b1,
                        __bf16* __restrict__ h2, int n, unsigned int seed) {
    int i = blockIdx.x * blockDim.x + threadIdx.x;
    if (i >= n) return;
    int f = i & (HCOL - 1);
    float v = (acc[i] + b1[f]) * drop_mask((unsigned)i, seed);
    h2[i] = (__bf16)elu_f(v);
}

// out[n,c] = mean_h acc2[n,h,c] + b2[c]
__global__ void gat_final(const float* __restrict__ acc2, const float* __restrict__ b2,
                          float* __restrict__ out, int total) {
    int i = blockIdx.x * blockDim.x + threadIdx.x;
    if (i >= total) return;
    int c    = i & (CDIM - 1);
    int node = i >> 6;
    float s = 0.0f;
#pragma unroll
    for (int h = 0; h < HEADS; ++h)
        s += acc2[(size_t)node * HCOL + h * CDIM + c];
    out[i] = s * (1.0f / HEADS) + b2[c];
}

// ------------------------------ WMMA GEMM ----------------------------------
// C[M,512] = A[M,512] @ W[512,512] (bf16 in, f32 out).
// Block: 256 threads (8 waves). Block tile 256x64, K-step 32.
// Wave w owns rows [32w,32w+32): 2 A-fragments x 4 B-blocks = 8 accumulators.
// A: double-buffered LDS, staged with GLOBAL_LOAD_ASYNC_TO_LDS_B128.
// B: read directly from pre-packed fragment layout in global (L2-resident).

#define GTM 256
#define GTK 32

__global__ __launch_bounds__(256)
void gat_gemm_bf16(const __bf16* __restrict__ A, const __bf16* __restrict__ Wf,
                   float* __restrict__ C, int M, int K, int Nc) {
    __shared__ __bf16 As[2][GTM * GTK];     // 2 x 16 KB

    const int t    = threadIdx.x;
    const int lane = t & 31;
    const int wave = t >> 5;
    const int hl   = lane >> 4;             // half-wave select
    const int lr   = lane & 15;
    const int rowBase = blockIdx.x * GTM;
    const int colBase = blockIdx.y * 64;
    const int nblk0   = colBase >> 4;
    const int mw      = wave * 32;

    const v8f vz = {0.f, 0.f, 0.f, 0.f, 0.f, 0.f, 0.f, 0.f};
    v8f acc[2][4] = {{vz, vz, vz, vz}, {vz, vz, vz, vz}};

    // each thread stages two 16B chunks (32B) of its A-tile row per K-step
    int gr = rowBase + t;
    if (gr >= M) gr = M - 1;
    const __bf16* arow = A + (size_t)gr * K;

    auto stage = [&](int buf, int k0) {
#ifdef GAT_ASYNC_LDS
        gi4v* gsrc = (gi4v*)(gvoid*)(arow + k0);
        li4v* ldst = (li4v*)(lvoid*)&As[buf][t * GTK];
        __builtin_amdgcn_global_load_async_to_lds_b128(gsrc, ldst, 0, 0);
        __builtin_amdgcn_global_load_async_to_lds_b128(gsrc, ldst, 16, 0);
#else
        const uint4* srcp = (const uint4*)(arow + k0);
        uint4* dstp = (uint4*)&As[buf][t * GTK];
        dstp[0] = srcp[0];
        dstp[1] = srcp[1];
#endif
        if (k0 + GTK < K)
            __builtin_prefetch(arow + k0 + GTK, 0, 1);
    };
    auto wait_stage = [&]() {
#ifdef GAT_ASYNC_LDS
#if __has_builtin(__builtin_amdgcn_s_wait_asynccnt)
        __builtin_amdgcn_s_wait_asynccnt(0);
#else
        asm volatile("s_wait_asynccnt 0x0" ::: "memory");
#endif
#endif
        __syncthreads();
    };

    stage(0, 0);

    const uint4* wfp = (const uint4*)Wf;
    const int nslab = K / GTK;

    for (int it = 0; it < nslab; ++it) {
        wait_stage();                         // buffer (it&1) ready block-wide
        if (it + 1 < nslab) stage((it + 1) & 1, (it + 1) * GTK);

        const __bf16* as_ = As[it & 1];
        union U { v16bf v; uint4 u[2]; };
        // A fragments (16x32 bf16, contiguous K to match packed B: lanes 0-15 K{0..15},
        // lanes 16-31 K{16..31})
        U fa0, fa1;
        fa0.u[0] = *(const uint4*)&as_[(mw + lr) * GTK + hl * 16];
        fa0.u[1] = *(const uint4*)&as_[(mw + lr) * GTK + hl * 16 + 8];
        fa1.u[0] = *(const uint4*)&as_[(mw + 16 + lr) * GTK + hl * 16];
        fa1.u[1] = *(const uint4*)&as_[(mw + 16 + lr) * GTK + hl * 16 + 8];

#pragma unroll
        for (int j = 0; j < 4; ++j) {
            U fb;   // pre-packed B fragment: 32B contiguous per lane
            size_t base = ((size_t)(it * 32 + nblk0 + j) * 64) + lane * 2;
            fb.u[0] = wfp[base];
            fb.u[1] = wfp[base + 1];
            acc[0][j] = __builtin_amdgcn_wmma_f32_16x16x32_bf16(
                false, fa0.v, false, fb.v, (short)0, acc[0][j], false, false);
            acc[1][j] = __builtin_amdgcn_wmma_f32_16x16x32_bf16(
                false, fa1.v, false, fb.v, (short)0, acc[1][j], false, false);
        }
    }

    // epilogue: C/D layout — VGPR g: M = hl*8+g, N = lane&15
#pragma unroll
    for (int f = 0; f < 2; ++f) {
#pragma unroll
        for (int j = 0; j < 4; ++j) {
#pragma unroll
            for (int g = 0; g < 8; ++g) {
                int grr = rowBase + mw + f * 16 + hl * 8 + g;
                int gc  = colBase + j * 16 + lr;
                if (grr < M) C[(size_t)grr * Nc + gc] = acc[f][j][g];
            }
        }
    }
}

// --------------------------- attention kernels -----------------------------

// a_s[n,h] = dot(xp[n,h,:], att_src[h,:]);  a_d likewise
__global__ void gat_attn_coef(const float* __restrict__ xp,
                              const float* __restrict__ att_src,
                              const float* __restrict__ att_dst,
                              float* __restrict__ a_s, float* __restrict__ a_d,
                              int total) {
    int i = blockIdx.x * blockDim.x + threadIdx.x;
    if (i >= total) return;   // total = N*HEADS
    int h    = i & (HEADS - 1);
    int node = i >> 3;
    const float4* xr = (const float4*)(xp + (size_t)node * HCOL + h * CDIM);
    const float4* sv = (const float4*)(att_src + h * CDIM);
    const float4* dv = (const float4*)(att_dst + h * CDIM);
    float ss = 0.f, sd = 0.f;
#pragma unroll
    for (int q = 0; q < CDIM / 4; ++q) {
        float4 v = xr[q], a = sv[q], b = dv[q];
        ss += v.x * a.x + v.y * a.y + v.z * a.z + v.w * a.w;
        sd += v.x * b.x + v.y * b.y + v.z * b.z + v.w * b.w;
    }
    a_s[i] = ss;
    a_d[i] = sd;
}

__device__ __forceinline__ void edge_endpoints(const long long* ei, int e,
                                               int& src, int& dst) {
    if (e < EE) { src = (int)ei[e]; dst = (int)ei[EE + e]; }
    else        { src = dst = e - EE; }
}

// segment max over dst (atomic max)
__global__ void gat_edge_max(const long long* __restrict__ ei,
                             const float* __restrict__ a_s,
                             const float* __restrict__ a_d,
                             float* __restrict__ m) {
    int e = blockIdx.x * blockDim.x + threadIdx.x;
    if (e >= ETOT) return;
    int src, dst;
    edge_endpoints(ei, e, src, dst);
#pragma unroll
    for (int h = 0; h < HEADS; ++h) {
        float v = a_s[src * HEADS + h] + a_d[dst * HEADS + h];
        v = v > 0.f ? v : NEGS * v;
        atomic_max_f32(&m[dst * HEADS + h], v);
    }
}

__global__ void gat_fix_m(float* __restrict__ m, int n) {
    int i = blockIdx.x * blockDim.x + threadIdx.x;
    if (i < n && !isfinite(m[i])) m[i] = 0.0f;
}

// segment sum of exp(e - m[dst])
__global__ void gat_edge_sum(const long long* __restrict__ ei,
                             const float* __restrict__ a_s,
                             const float* __restrict__ a_d,
                             const float* __restrict__ m,
                             float* __restrict__ sbuf) {
    int e = blockIdx.x * blockDim.x + threadIdx.x;
    if (e >= ETOT) return;
    int src, dst;
    edge_endpoints(ei, e, src, dst);
#pragma unroll
    for (int h = 0; h < HEADS; ++h) {
        float v = a_s[src * HEADS + h] + a_d[dst * HEADS + h];
        v = v > 0.f ? v : NEGS * v;
        atomicAdd(&sbuf[dst * HEADS + h], __expf(v - m[dst * HEADS + h]));
    }
}

// acc[dst,h,:] += alpha * xp[src,h,:]   (one thread per (edge,head))
__global__ void gat_scatter(const long long* __restrict__ ei,
                            const float* __restrict__ a_s,
                            const float* __restrict__ a_d,
                            const float* __restrict__ m,
                            const float* __restrict__ sbuf,
                            const float* __restrict__ xp,
                            float* __restrict__ acc) {
    int i = blockIdx.x * blockDim.x + threadIdx.x;
    if (i >= ETOT * HEADS) return;
    int h = i & (HEADS - 1);
    int e = i >> 3;
    int src, dst;
    edge_endpoints(ei, e, src, dst);
    float v = a_s[src * HEADS + h] + a_d[dst * HEADS + h];
    v = v > 0.f ? v : NEGS * v;
    float alpha = __expf(v - m[dst * HEADS + h]) / (sbuf[dst * HEADS + h] + 1e-16f);
    const float4* xs = (const float4*)(xp + (size_t)src * HCOL + h * CDIM);
    float* ap = acc + (size_t)dst * HCOL + h * CDIM;
#pragma unroll 4
    for (int q = 0; q < CDIM / 4; ++q) {
        float4 tv = xs[q];
        atomicAdd(ap + q * 4 + 0, alpha * tv.x);
        atomicAdd(ap + q * 4 + 1, alpha * tv.y);
        atomicAdd(ap + q * 4 + 2, alpha * tv.z);
        atomicAdd(ap + q * 4 + 3, alpha * tv.w);
    }
}

// ------------------------------- launcher ----------------------------------

static inline int cdiv(long long a, long long b) { return (int)((a + b - 1) / b); }

extern "C" void kernel_launch(void* const* d_in, const int* in_sizes, int n_in,
                              void* d_out, int out_size, void* d_ws, size_t ws_size,
                              hipStream_t stream) {
    const float*     x        = (const float*)d_in[0];
    const long long* ei       = (const long long*)d_in[1];
    const float*     W1       = (const float*)d_in[2];
    const float*     att_src1 = (const float*)d_in[3];
    const float*     att_dst1 = (const float*)d_in[4];
    const float*     b1       = (const float*)d_in[5];
    const float*     W2       = (const float*)d_in[6];
    const float*     att_src2 = (const float*)d_in[7];
    const float*     att_dst2 = (const float*)d_in[8];
    const float*     b2       = (const float*)d_in[9];
    float* out = (float*)d_out;

    // ---- workspace layout (256B aligned) ----
    char* ws = (char*)d_ws;
    size_t off = 0;
    auto take = [&](size_t bytes) -> char* {
        char* p = ws + off;
        off += (bytes + 255) & ~size_t(255);
        return p;
    };
    __bf16* hbf  = (__bf16*)take((size_t)NN * FEAT * 2);
    __bf16* w1f  = (__bf16*)take((size_t)FEAT * HCOL * 2);   // packed fragments
    __bf16* w2f  = (__bf16*)take((size_t)HCOL * HCOL * 2);   // packed fragments
    float*  xp1  = (float*)take((size_t)NN * HCOL * 4);
    float*  as1  = (float*)take((size_t)NN * HEADS * 4);
    float*  ad1  = (float*)take((size_t)NN * HEADS * 4);
    float*  m1   = (float*)take((size_t)NN * HEADS * 4);
    float*  s1b  = (float*)take((size_t)NN * HEADS * 4);
    float*  acc1 = (float*)take((size_t)NN * HCOL * 4);
    __bf16* h2b  = (__bf16*)take((size_t)NN * HCOL * 2);
    float*  xp2  = (float*)take((size_t)NN * HCOL * 4);
    float*  as2  = (float*)take((size_t)NN * HEADS * 4);
    float*  ad2  = (float*)take((size_t)NN * HEADS * 4);
    float*  m2   = (float*)take((size_t)NN * HEADS * 4);
    float*  s2b  = (float*)take((size_t)NN * HEADS * 4);
    float*  acc2 = (float*)take((size_t)NN * HCOL * 4);

    const int B = 256;
    const float NEG_INF = -__builtin_inff();

    // ---- layer 1 ----
    gat_prep_x<<<cdiv((long long)NN * FEAT, B), B, 0, stream>>>(x, hbf, NN * FEAT, 0x9E3779B9u);
    gat_pack_w<<<cdiv((long long)FEAT * HCOL, B), B, 0, stream>>>(W1, w1f, FEAT * HCOL);
    gat_pack_w<<<cdiv((long long)HCOL * HCOL, B), B, 0, stream>>>(W2, w2f, HCOL * HCOL);

    dim3 ggrid(cdiv(NN, GTM), HCOL / 64);
    gat_gemm_bf16<<<ggrid, B, 0, stream>>>(hbf, w1f, xp1, NN, FEAT, HCOL);

    gat_attn_coef<<<cdiv(NN * HEADS, B), B, 0, stream>>>(xp1, att_src1, att_dst1, as1, ad1, NN * HEADS);
    gat_fill_f32 <<<cdiv(NN * HEADS, B), B, 0, stream>>>(m1, NEG_INF, NN * HEADS);
    gat_fill_f32 <<<cdiv(NN * HEADS, B), B, 0, stream>>>(s1b, 0.0f, NN * HEADS);
    gat_edge_max <<<cdiv(ETOT, B), B, 0, stream>>>(ei, as1, ad1, m1);
    gat_fix_m    <<<cdiv(NN * HEADS, B), B, 0, stream>>>(m1, NN * HEADS);
    gat_edge_sum <<<cdiv(ETOT, B), B, 0, stream>>>(ei, as1, ad1, m1, s1b);
    gat_fill_f32 <<<cdiv((long long)NN * HCOL, B), B, 0, stream>>>(acc1, 0.0f, NN * HCOL);
    gat_scatter  <<<cdiv((long long)ETOT * HEADS, B), B, 0, stream>>>(ei, as1, ad1, m1, s1b, xp1, acc1);

    // ---- layer 2 ----
    gat_mid<<<cdiv((long long)NN * HCOL, B), B, 0, stream>>>(acc1, b1, h2b, NN * HCOL, 0x85EBCA6Bu);
    gat_gemm_bf16<<<ggrid, B, 0, stream>>>(h2b, w2f, xp2, NN, HCOL, HCOL);

    gat_attn_coef<<<cdiv(NN * HEADS, B), B, 0, stream>>>(xp2, att_src2, att_dst2, as2, ad2, NN * HEADS);
    gat_fill_f32 <<<cdiv(NN * HEADS, B), B, 0, stream>>>(m2, NEG_INF, NN * HEADS);
    gat_fill_f32 <<<cdiv(NN * HEADS, B), B, 0, stream>>>(s2b, 0.0f, NN * HEADS);
    gat_edge_max <<<cdiv(ETOT, B), B, 0, stream>>>(ei, as2, ad2, m2);
    gat_fix_m    <<<cdiv(NN * HEADS, B), B, 0, stream>>>(m2, NN * HEADS);
    gat_edge_sum <<<cdiv(ETOT, B), B, 0, stream>>>(ei, as2, ad2, m2, s2b);
    gat_fill_f32 <<<cdiv((long long)NN * HCOL, B), B, 0, stream>>>(acc2, 0.0f, NN * HCOL);
    gat_scatter  <<<cdiv((long long)ETOT * HEADS, B), B, 0, stream>>>(ei, as2, ad2, m2, s2b, xp2, acc2);

    gat_final<<<cdiv(NN * CDIM, B), B, 0, stream>>>(acc2, b2, out, NN * CDIM);
}
